// SPIDERMouse_28819230556956
// MI455X (gfx1250) — matrix operand order repaired and checked by reading, not compiled, not run on hardware
//
#include <hip/hip_runtime.h>
#include <hip/hip_bf16.h>
#include <stdint.h>

#define NN   50000
#define EE   500000
#define EPSBN 1e-5f

typedef __attribute__((ext_vector_type(16))) __bf16 v16bf;
typedef __attribute__((ext_vector_type(8)))  __bf16 v8bf;
typedef __attribute__((ext_vector_type(8)))  float  v8f;

__device__ __forceinline__ void atomicMaxF(float* addr, float val) {
  unsigned* ua = (unsigned*)addr;
  unsigned old = *ua;
  while (__uint_as_float(old) < val) {
    unsigned assumed = old;
    old = atomicCAS(ua, assumed, __float_as_uint(val));
    if (old == assumed) break;
  }
}

// ---------------- elementwise utility kernels ----------------
__global__ void spd_fill(float* __restrict__ p, float v, long long n) {
  long long i = (long long)blockIdx.x * blockDim.x + threadIdx.x;
  long long st = (long long)gridDim.x * blockDim.x;
  for (; i < n; i += st) p[i] = v;
}

__global__ void spd_add(float* __restrict__ a, const float* __restrict__ b, long long n) {
  long long i = (long long)blockIdx.x * blockDim.x + threadIdx.x;
  long long st = (long long)gridDim.x * blockDim.x;
  for (; i < n; i += st) a[i] += b[i];
}

__global__ void spd_addbias_mod(float* __restrict__ a, const float* __restrict__ bias,
                                int C, long long n) {
  long long i = (long long)blockIdx.x * blockDim.x + threadIdx.x;
  long long st = (long long)gridDim.x * blockDim.x;
  for (; i < n; i += st) a[i] += bias[(int)(i % C)];
}

// ---------------- weight packing into WMMA B lane layout ----------------
// B (KxNc, row-major f32)  ->  tiles[kt][nt][lane 0..31][e 0..15] bf16
// lane l: column n = nt*16 + (l&15); k = kt*32 + (l>>4)*8 + e + (e>=8 ? 8 : 0)
__global__ void spd_pack_b(const float* __restrict__ B, __bf16* __restrict__ out,
                           int K, int Nc, int NT) {
  int t = blockIdx.x * blockDim.x + threadIdx.x;
  int KT = (K + 31) >> 5;
  int total = KT * NT * 32;
  if (t >= total) return;
  int lane = t & 31;
  int tile = t >> 5;
  int tn = tile % NT;
  int kt = tile / NT;
  int n  = tn * 16 + (lane & 15);
  int kb = kt * 32 + (lane >> 4) * 8;
  __bf16* o = out + (size_t)t * 16;
#pragma unroll
  for (int e = 0; e < 16; ++e) {
    int k = kb + e + ((e >> 3) << 3);
    float v = (k < K && n < Nc) ? B[(size_t)k * Nc + n] : 0.f;
    o[e] = (__bf16)v;
  }
}

// ---------------- generic WMMA GEMM:  C = gather(A) @ B + bias ----------------
// 128 threads = 4 waves per block; each wave owns one 16x16 output tile.
// blockIdx.x covers 4 consecutive M tiles, blockIdx.y = N tile.
// Optional row indirection: row = idx[m*2 + idxOff] (edge endpoint gather fused).
__global__ __launch_bounds__(128)
void spd_gemm_wmma(const float* __restrict__ A, int lda, int acol0,
                   const int* __restrict__ idx, int idxOff,
                   const __bf16* __restrict__ Bp,
                   const float* __restrict__ bias,
                   float* __restrict__ C, int ldc, int ccol0,
                   int M, int K, int Nc) {
  const int lane = threadIdx.x & 31;
  const int wave = threadIdx.x >> 5;
  const int tm = blockIdx.x * 4 + wave;
  if (tm * 16 >= M) return;                 // uniform per wave: EXEC stays all-ones below
  const int tn = blockIdx.y;
  const int NT = gridDim.y;
  const int grp = lane >> 4;

  int m = tm * 16 + (lane & 15);
  int mc = (m < M) ? m : (M - 1);
  long long arow = idx ? (long long)idx[(size_t)mc * 2 + idxOff] : (long long)mc;
  const float* Ar = A + (size_t)arow * lda + acol0 + grp * 8;

  const v16bf* Bt = (const v16bf*)Bp;
  v8f acc = {0.f, 0.f, 0.f, 0.f, 0.f, 0.f, 0.f, 0.f};

  const int KFULL = K >> 5;                 // uniform full-tile count
#pragma unroll 2
  for (int kt = 0; kt < KFULL; ++kt) {
    const float* Ak = Ar + kt * 32;
    v8f x0 = *(const v8f*)(Ak);             // 32B aligned (all strides mult. of 8 floats)
    v8f x1 = *(const v8f*)(Ak + 16);
    __builtin_prefetch(Ak + 32, 0, 1);      // speculative global_prefetch_b8
    v8bf lo = __builtin_convertvector(x0, v8bf);   // v_cvt_pk_bf16_f32 x4
    v8bf hi = __builtin_convertvector(x1, v8bf);
    v16bf a = __builtin_shufflevector(lo, hi, 0,1,2,3,4,5,6,7,8,9,10,11,12,13,14,15);
    v16bf b = Bt[((size_t)kt * NT + tn) * 32 + lane];
    acc = __builtin_amdgcn_wmma_f32_16x16x32_bf16(false, a, false, b,
                                                  (short)0, acc, false, false);
  }
  if (K & 31) {                             // uniform remainder tile (only K=1 GEMMs)
    int kb = KFULL * 32 + grp * 8;
    v8f x0, x1;
#pragma unroll
    for (int e = 0; e < 8; ++e) { int k = kb + e;      x0[e] = (k < K) ? (A + (size_t)arow * lda + acol0)[k] : 0.f; }
#pragma unroll
    for (int e = 0; e < 8; ++e) { int k = kb + 16 + e; x1[e] = (k < K) ? (A + (size_t)arow * lda + acol0)[k] : 0.f; }
    v8bf lo = __builtin_convertvector(x0, v8bf);
    v8bf hi = __builtin_convertvector(x1, v8bf);
    v16bf a = __builtin_shufflevector(lo, hi, 0,1,2,3,4,5,6,7,8,9,10,11,12,13,14,15);
    v16bf b = Bt[((size_t)KFULL * NT + tn) * 32 + lane];
    acc = __builtin_amdgcn_wmma_f32_16x16x32_bf16(false, a, false, b,
                                                  (short)0, acc, false, false);
  }

  int n = tn * 16 + (lane & 15);
  if (n >= Nc) return;
  float bb = bias ? bias[n] : 0.f;
  int mb = tm * 16 + grp * 8;
#pragma unroll
  for (int r = 0; r < 8; ++r) {
    int mm = mb + r;
    if (mm < M) C[(size_t)mm * ldc + ccol0 + n] = acc[r] + bb;
  }
}

// ---------------- batch-norm (axis 0) ----------------
__global__ void spd_bn_stats(const float* __restrict__ Y, int ld, int col0,
                             int M, int C, float* __restrict__ stats) {
  int c = blockIdx.x;           // one block per column
  __shared__ float ss[256];
  __shared__ float sq[256];
  float s = 0.f, q = 0.f;
  for (int r = threadIdx.x; r < M; r += blockDim.x) {
    float v = Y[(size_t)r * ld + col0 + c];
    s += v; q += v * v;
  }
  ss[threadIdx.x] = s; sq[threadIdx.x] = q;
  __syncthreads();
  for (int o = 128; o > 0; o >>= 1) {
    if (threadIdx.x < o) { ss[threadIdx.x] += ss[threadIdx.x + o]; sq[threadIdx.x] += sq[threadIdx.x + o]; }
    __syncthreads();
  }
  if (threadIdx.x == 0) {
    float mean = ss[0] / (float)M;
    float var  = sq[0] / (float)M - mean * mean;
    stats[c]     = mean;
    stats[C + c] = rsqrtf(var + EPSBN);
  }
}

__global__ void spd_bn_apply(float* __restrict__ Y, int ld, int col0,
                             long long M, int C, const float* __restrict__ stats,
                             const float* __restrict__ g, const float* __restrict__ be,
                             float slope) {
  long long tot = M * (long long)C;
  long long i = (long long)blockIdx.x * blockDim.x + threadIdx.x;
  long long st = (long long)gridDim.x * blockDim.x;
  for (; i < tot; i += st) {
    long long r = i / C;
    int c = (int)(i % C);
    size_t o = (size_t)r * ld + col0 + c;
    float v = Y[o];
    v = (v - stats[c]) * stats[C + c] * g[c] + be[c];
    Y[o] = (v >= 0.f) ? v : slope * v;
  }
}

// ---------------- GAT attention ----------------
// wave per (edge-pair j in [0,2E), head h): score + fused segment-max
__global__ void spd_score(const float* __restrict__ xl, const float* __restrict__ xr,
                          const float* __restrict__ att, const int* __restrict__ ei,
                          int E, float* __restrict__ score, float* __restrict__ mbuf) {
  int wid = (blockIdx.x * blockDim.x + threadIdx.x) >> 5;
  int lane = threadIdx.x & 31;
  if (wid >= 2 * E * 4) return;
  int h = wid & 3;
  int j = wid >> 2;
  int e = (j < E) ? j : (j - E);
  int src = ei[2 * e], dst = ei[2 * e + 1];
  const float* pl = xl + (size_t)src * 256 + h * 64;
  const float* pr = xr + (size_t)dst * 256 + h * 64;
  const float* pa = att + h * 64;
  float s = 0.f;
#pragma unroll
  for (int c0 = 0; c0 < 64; c0 += 32) {
    int c = c0 + lane;
    float v = pl[c] + pr[c];
    v = (v >= 0.f) ? v : 0.2f * v;
    s += v * pa[c];
  }
  for (int o = 16; o > 0; o >>= 1) s += __shfl_down(s, o, 32);
  if (lane == 0) {
    score[(size_t)j * 4 + h] = s;
    atomicMaxF(&mbuf[(size_t)dst * 4 + h], s);
  }
}

__global__ void spd_ex_denom(float* __restrict__ score, const float* __restrict__ mbuf,
                             float* __restrict__ denom, const int* __restrict__ ei, int E) {
  int t = blockIdx.x * blockDim.x + threadIdx.x;
  if (t >= 2 * E * 4) return;
  int h = t & 3;
  int j = t >> 2;
  int e = (j < E) ? j : (j - E);
  int dst = ei[2 * e + 1];
  float ex = __expf(score[t] - mbuf[(size_t)dst * 4 + h]);
  score[t] = ex;
  atomicAdd(&denom[(size_t)dst * 4 + h], ex);
}

__global__ void spd_alpha(const float* __restrict__ ex, const float* __restrict__ denom,
                          const int* __restrict__ ei, int E, float* __restrict__ alpha_out) {
  int t = blockIdx.x * blockDim.x + threadIdx.x;
  if (t >= 2 * E * 4) return;
  int h = t & 3;
  int j = t >> 2;
  int e = (j < E) ? j : (j - E);
  int dst = ei[2 * e + 1];
  alpha_out[t] = ex[t] / denom[(size_t)dst * 4 + h];
}

// wave per (j,h): h1[dst, h*64+c] += alpha * xl[src, h*64+c]
__global__ void spd_h1_agg(const float* __restrict__ alpha, const float* __restrict__ xl,
                           const int* __restrict__ ei, int E, float* __restrict__ h1) {
  int wid = (blockIdx.x * blockDim.x + threadIdx.x) >> 5;
  int lane = threadIdx.x & 31;
  if (wid >= 2 * E * 4) return;
  int h = wid & 3;
  int j = wid >> 2;
  int e = (j < E) ? j : (j - E);
  int src = ei[2 * e], dst = ei[2 * e + 1];
  float a = alpha[(size_t)j * 4 + h];
  const float* px = xl + (size_t)src * 256 + h * 64;
  float* ph = h1 + (size_t)dst * 256 + h * 64;
  atomicAdd(&ph[lane],      a * px[lane]);
  atomicAdd(&ph[lane + 32], a * px[lane + 32]);
}

// ---------------- final 32->1 + sigmoid ----------------
__global__ void spd_fc3_sigmoid(const float* __restrict__ X, const float* __restrict__ W3,
                                const float* __restrict__ b3, int E, float* __restrict__ out) {
  int e = blockIdx.x * blockDim.x + threadIdx.x;
  if (e >= E) return;
  const float* x = X + (size_t)e * 32;
  float s = b3[0];
#pragma unroll
  for (int c = 0; c < 32; ++c) s += x[c] * W3[c];
  out[e] = 1.f / (1.f + __expf(-s));
}

// ====================== host orchestration ======================
extern "C" void kernel_launch(void* const* d_in, const int* in_sizes, int n_in,
                              void* d_out, int out_size, void* d_ws, size_t ws_size,
                              hipStream_t stream) {
  (void)in_sizes; (void)n_in; (void)out_size; (void)ws_size;
#define PF(i) ((const float*)d_in[(i)])
  const float* graph = PF(0);
  const float* inter = PF(1);
  const int*   ei    = (const int*)d_in[2];
  // submodel param blocks: W1,b1,g1,be1,W2,b2,g2,be2
  const int G = 3, Pp = 11, Ll = 19, COP = 27, COL = 35;
  const float *gatWl = PF(43), *gatWr = PF(44), *gatAtt = PF(45), *gatBias = PF(46);
  const float *phiW = PF(47), *phiB = PF(48), *phiG = PF(49), *phiBe = PF(50);
  const int RHO = 51, FC = 59;
  const float *fcW3 = PF(67), *fcB3 = PF(68);

  float* out = (float*)d_out;              // [0,E) preds ; [E, E+2E*4) alpha
  const int N = NN, E = EE;

  // ---- bump allocator over workspace ----
  char* base = (char*)d_ws;
  size_t off = 0;
  auto alloc = [&](size_t bytes) -> void* {
    void* p = base + off;
    off = (off + bytes + 255) & ~(size_t)255;
    return p;
  };

  auto packB = [&](const float* W, int K, int Nc) -> const __bf16* {
    int KT = (K + 31) / 32, NT = (Nc + 15) / 16;
    __bf16* o = (__bf16*)alloc((size_t)KT * NT * 512 * 2);
    int total = KT * NT * 32;
    spd_pack_b<<<(total + 255) / 256, 256, 0, stream>>>(W, o, K, Nc, NT);
    return o;
  };

  // packed weights (deterministic, re-done each call)
  const __bf16* Pg1  = packB(PF(G),      1024, 32);
  const __bf16* Pg2  = packB(PF(G + 4),    32, 64);
  const __bf16* Pp1  = packB(PF(Pp),      512, 32);
  const __bf16* Pp2  = packB(PF(Pp + 4),   32, 64);
  const __bf16* Pl1  = packB(PF(Ll),      256, 32);
  const __bf16* Pl2  = packB(PF(Ll + 4),   32, 64);
  const __bf16* PWl  = packB(gatWl,       192, 256);
  const __bf16* PWr  = packB(gatWr,       192, 256);
  const __bf16* Pphi = packB(phiW,        256, 128);
  const __bf16* Pr1  = packB(PF(RHO),     128, 128);
  const __bf16* Pr2  = packB(PF(RHO + 4), 128, 64);
  const __bf16* Pcp1 = packB(PF(COP),       1, 32);
  const __bf16* Pcp2 = packB(PF(COP + 4),  32, 64);
  const __bf16* Pcl1 = packB(PF(COL),       1, 32);
  const __bf16* Pcl2 = packB(PF(COL + 4),  32, 64);
  const __bf16* Pf1  = packB(PF(FC),      192, 64);
  const __bf16* Pf2  = packB(PF(FC + 4),   64, 32);

  // workspace buffers (floats)
  float* stats = (float*)alloc(512 * sizeof(float));
  float* y1    = (float*)alloc((size_t)N * 32  * sizeof(float));
  float* gm    = (float*)alloc((size_t)N * 192 * sizeof(float));
  float* xl    = (float*)alloc((size_t)N * 256 * sizeof(float));
  float* xr    = (float*)alloc((size_t)N * 256 * sizeof(float));
  float* h1    = (float*)alloc((size_t)N * 256 * sizeof(float));
  float* score = (float*)alloc((size_t)2 * E * 4 * sizeof(float));
  float* mbuf  = (float*)alloc((size_t)N * 4 * sizeof(float));
  float* denom = (float*)alloc((size_t)N * 4 * sizeof(float));
  float* yE    = (float*)alloc((size_t)E * 32  * sizeof(float));
  float* bufA  = (float*)alloc((size_t)E * 192 * sizeof(float));  // phiA -> cat(E x 192)
  float* bufB  = (float*)alloc((size_t)E * 160 * sizeof(float));  // phiB/rho1/fc1 + fc2
  float* bufB2 = bufB + (size_t)E * 64;

  auto gemm = [&](const float* A, int lda, int acol0, const int* gidx, int gioff,
                  const __bf16* Bp, const float* bias, float* C, int ldc, int ccol0,
                  int M, int K, int Nc) {
    int MT = (M + 15) / 16;
    dim3 g((MT + 3) / 4, (Nc + 15) / 16);
    spd_gemm_wmma<<<g, 128, 0, stream>>>(A, lda, acol0, gidx, gioff, Bp, bias,
                                         C, ldc, ccol0, M, K, Nc);
  };
  auto bn = [&](float* Y, int ld, int col0, int M, int C,
                const float* g, const float* be) {
    spd_bn_stats<<<C, 256, 0, stream>>>(Y, ld, col0, M, C, stats);
    spd_bn_apply<<<4096, 256, 0, stream>>>(Y, ld, col0, (long long)M, C, stats, g, be, 0.01f);
  };
  auto submodel = [&](const float* A, int lda, int acol0, int fin, int M,
                      const __bf16* PB1, const __bf16* PB2, int pbase,
                      float* hid, int ldh, float* outY, int ldo, int ocol0) {
    gemm(A, lda, acol0, nullptr, 0, PB1, PF(pbase + 1), hid, ldh, 0, M, fin, 32);
    bn(hid, ldh, 0, M, 32, PF(pbase + 2), PF(pbase + 3));
    gemm(hid, ldh, 0, nullptr, 0, PB2, PF(pbase + 5), outY, ldo, ocol0, M, 32, 64);
    bn(outY, ldo, ocol0, M, 64, PF(pbase + 6), PF(pbase + 7));
  };

  // ---- init accumulators ----
  spd_fill<<<4096, 256, 0, stream>>>(h1, 0.f, (long long)N * 256);
  spd_fill<<<256, 256, 0, stream>>>(mbuf, -3.0e38f, (long long)N * 4);
  spd_fill<<<256, 256, 0, stream>>>(denom, 0.f, (long long)N * 4);

  // ---- node submodels -> gm (N x 192) ----
  submodel(graph, 1792, 0,    1024, N, Pg1, Pg2, G,  y1, 32, gm, 192, 0);
  submodel(graph, 1792, 1024,  512, N, Pp1, Pp2, Pp, y1, 32, gm, 192, 64);
  submodel(graph, 1792, 1536,  256, N, Pl1, Pl2, Ll, y1, 32, gm, 192, 128);

  // ---- GAT projections ----
  gemm(gm, 192, 0, nullptr, 0, PWl, nullptr, xl, 256, 0, N, 192, 256);
  gemm(gm, 192, 0, nullptr, 0, PWr, nullptr, xr, 256, 0, N, 192, 256);

  // ---- attention: score + segmax, exp+denom, alpha (-> d_out), h1 segsum ----
  int waves = 2 * E * 4;
  spd_score<<<(waves * 32 + 255) / 256, 256, 0, stream>>>(xl, xr, gatAtt, ei, E, score, mbuf);
  spd_ex_denom<<<(waves + 255) / 256, 256, 0, stream>>>(score, mbuf, denom, ei, E);
  spd_alpha<<<(waves + 255) / 256, 256, 0, stream>>>(score, denom, ei, E, out + E);
  spd_h1_agg<<<(waves * 32 + 255) / 256, 256, 0, stream>>>(out + E, xl, ei, E, h1);
  spd_addbias_mod<<<4096, 256, 0, stream>>>(h1, gatBias, 256, (long long)N * 256);

  // ---- phi on both endpoints (gather fused into GEMM A-loader) ----
  gemm(h1, 256, 0, ei, 0, Pphi, phiB, bufA, 128, 0, E, 256, 128);
  bn(bufA, 128, 0, E, 128, phiG, phiBe);
  gemm(h1, 256, 0, ei, 1, Pphi, phiB, bufB, 128, 0, E, 256, 128);
  bn(bufB, 128, 0, E, 128, phiG, phiBe);
  spd_add<<<4096, 256, 0, stream>>>(bufA, bufB, (long long)E * 128);   // preds1

  // ---- rho ----
  gemm(bufA, 128, 0, nullptr, 0, Pr1, PF(RHO + 1), bufB, 128, 0, E, 128, 128);
  bn(bufB, 128, 0, E, 128, PF(RHO + 2), PF(RHO + 3));
  gemm(bufB, 128, 0, nullptr, 0, Pr2, PF(RHO + 5), bufA, 192, 0, E, 128, 64);  // cat cols 0..63
  bn(bufA, 192, 0, E, 64, PF(RHO + 6), PF(RHO + 7));

  // ---- co_p / co_l scalar-input submodels -> cat cols 64..127 / 128..191 ----
  submodel(inter, 2, 0, 1, E, Pcp1, Pcp2, COP, yE, 32, bufA, 192, 64);
  submodel(inter, 2, 1, 1, E, Pcl1, Pcl2, COL, yE, 32, bufA, 192, 128);

  // ---- fc head ----
  gemm(bufA, 192, 0, nullptr, 0, Pf1, PF(FC + 1), bufB, 64, 0, E, 192, 64);
  bn(bufB, 64, 0, E, 64, PF(FC + 2), PF(FC + 3));
  gemm(bufB, 64, 0, nullptr, 0, Pf2, PF(FC + 5), bufB2, 32, 0, E, 64, 32);
  bn(bufB2, 32, 0, E, 32, PF(FC + 6), PF(FC + 7));
  spd_fc3_sigmoid<<<(E + 255) / 256, 256, 0, stream>>>(bufB2, fcW3, fcB3, E, out);
#undef PF
}
